// Features_70806830841880
// MI455X (gfx1250) — matrix operand order repaired and checked by reading, not compiled, not run on hardware
//
#include <hip/hip_runtime.h>

#define NP   3136      // patch rows (56*56)
#define NL   16384     // patch_lib rows
#define DIM  768
#define IMG  224
#define FM   56

typedef __attribute__((ext_vector_type(16))) __bf16 v16bf;
typedef __attribute__((ext_vector_type(8)))  __bf16 v8bf;
typedef __attribute__((ext_vector_type(8)))  float  v8f;

// ---------- helpers ----------
static __device__ __forceinline__ unsigned short f2bf(float f) {
    unsigned u = __float_as_uint(f);
    unsigned r = u + 0x7FFFu + ((u >> 16) & 1u);   // round-to-nearest-even
    return (unsigned short)(r >> 16);
}
static __device__ __forceinline__ float bf2f(unsigned short h) {
    return __uint_as_float(((unsigned)h) << 16);
}
static __device__ __forceinline__ int iclampi(int v, int lo, int hi) {
    return v < lo ? lo : (v > hi ? hi : v);
}
static __device__ __forceinline__ int mirror(int i, int n) {
    if (i < 0) i = -i;
    if (i >= n) i = 2 * n - 2 - i;
    return i;
}
static __device__ __forceinline__ v8f wmma_bf16(v16bf a, v16bf b, v8f c) {
    return __builtin_amdgcn_wmma_f32_16x16x32_bf16(false, a, false, b, (short)0, c, false, false);
}

// ---------- 1) fp32 -> (bf16 hi, bf16 lo) planes + row squared norms ----------
__global__ __launch_bounds__(256) void prep_kernel(
    const float* __restrict__ x, const float* __restrict__ mean,
    const float* __restrict__ std_, unsigned short* __restrict__ hi,
    unsigned short* __restrict__ lo, float* __restrict__ nrm, int normalize)
{
    int row = blockIdx.x;
    float mu = normalize ? mean[0] : 0.0f;
    float sd = normalize ? std_[0] : 1.0f;
    const float* xr = x + (size_t)row * DIM;
    size_t base = (size_t)row * DIM;
    float acc = 0.f;
    for (int i = threadIdx.x; i < DIM; i += 256) {
        float v = (xr[i] - mu) / sd;
        unsigned short h = f2bf(v);
        unsigned short l = f2bf(v - bf2f(h));
        hi[base + i] = h;
        lo[base + i] = l;
        acc += v * v;
    }
    for (int off = 16; off > 0; off >>= 1) acc += __shfl_xor(acc, off, 32);
    __shared__ float red[8];
    if ((threadIdx.x & 31) == 0) red[threadIdx.x >> 5] = acc;
    __syncthreads();
    if (threadIdx.x == 0) {
        float s = 0.f;
        for (int i = 0; i < 8; ++i) s += red[i];
        nrm[row] = s;
    }
}

// ---------- 2) init packed min buffer ----------
__global__ void init_minpack(unsigned long long* mp, int n) {
    int i = blockIdx.x * 256 + threadIdx.x;
    if (i < n) mp[i] = ~0ull;
}

// ---------- 3) fused cdist GEMM (bf16 hi/lo split WMMA) + row min/argmin ----------
// block = 256 thr = 8 waves; wave = 32 rows x 64 cols (2 A frag pairs x 4 B tile pairs,
// 8 accumulators, 24 WMMA : 24 b128 loads per K=32 step -> 1:1 load:math ratio)
// block tile = 64 rows x 256 cols; grid = (NL/256, NP/64)
__global__ __launch_bounds__(256) void gemm_min_kernel(
    const unsigned short* __restrict__ Ahi_, const unsigned short* __restrict__ Alo_,
    const unsigned short* __restrict__ Bhi_, const unsigned short* __restrict__ Blo_,
    const float* __restrict__ a2, const float* __restrict__ b2,
    unsigned long long* __restrict__ minpack)
{
    __shared__ unsigned long long smin[64];
    const __bf16* Ahi = (const __bf16*)Ahi_;
    const __bf16* Alo = (const __bf16*)Alo_;
    const __bf16* Bhi = (const __bf16*)Bhi_;
    const __bf16* Blo = (const __bf16*)Blo_;

    int tid  = threadIdx.x;
    int lane = tid & 31;
    int wave = tid >> 5;
    int rowbase = blockIdx.y * 64 + (wave >> 2) * 32;
    int colbase = blockIdx.x * 256 + (wave & 3) * 64;

    if (tid < 64) smin[tid] = ~0ull;
    __syncthreads();

    int m  = lane & 15;
    int kh = lane >> 4;                      // which K-half this lane holds

    // A fragment sources (16 contiguous K at kh*16, matching B)
    const __bf16* a0_hi = Ahi + (size_t)(rowbase + m) * DIM;
    const __bf16* a0_lo = Alo + (size_t)(rowbase + m) * DIM;
    const __bf16* a1_hi = a0_hi + 16 * DIM;
    const __bf16* a1_lo = a0_lo + 16 * DIM;
    // B fragment sources (16 contiguous K at kh*16, same order as A)
    const __bf16* bhp = Bhi + (size_t)(colbase + m) * DIM + kh * 16;
    const __bf16* blp = Blo + (size_t)(colbase + m) * DIM + kh * 16;

    v8f acc[2][4];
    #pragma unroll
    for (int i = 0; i < 2; ++i)
        #pragma unroll
        for (int j = 0; j < 4; ++j) acc[i][j] = (v8f){};

    for (int k0 = 0; k0 < DIM; k0 += 32) {
        union { v16bf v; v8bf h[2]; } ah0, ah1, al0, al1;
        ah0.h[0] = *(const v8bf*)(a0_hi + k0 + kh * 16);
        ah0.h[1] = *(const v8bf*)(a0_hi + k0 + 8 + kh * 16);
        ah1.h[0] = *(const v8bf*)(a1_hi + k0 + kh * 16);
        ah1.h[1] = *(const v8bf*)(a1_hi + k0 + 8 + kh * 16);
        al0.h[0] = *(const v8bf*)(a0_lo + k0 + kh * 16);
        al0.h[1] = *(const v8bf*)(a0_lo + k0 + 8 + kh * 16);
        al1.h[0] = *(const v8bf*)(a1_lo + k0 + kh * 16);
        al1.h[1] = *(const v8bf*)(a1_lo + k0 + 8 + kh * 16);
        #pragma unroll
        for (int ct = 0; ct < 4; ++ct) {
            v16bf bh = *(const v16bf*)(bhp + (size_t)ct * 16 * DIM + k0);
            v16bf bl = *(const v16bf*)(blp + (size_t)ct * 16 * DIM + k0);
            // dot ~= hi*hi + hi*lo + lo*hi  (fp32-accurate split product)
            acc[0][ct] = wmma_bf16(ah0.v, bh, acc[0][ct]);
            acc[0][ct] = wmma_bf16(ah0.v, bl, acc[0][ct]);
            acc[0][ct] = wmma_bf16(al0.v, bh, acc[0][ct]);
            acc[1][ct] = wmma_bf16(ah1.v, bh, acc[1][ct]);
            acc[1][ct] = wmma_bf16(ah1.v, bl, acc[1][ct]);
            acc[1][ct] = wmma_bf16(al1.v, bh, acc[1][ct]);
        }
    }

    // epilogue: d2 = |a|^2 + |b|^2 - 2*dot ; pack (d2bits, col); per-lane min over
    // 4 col tiles, then 16-lane shuffle min, then LDS + global packed atomic min
    float b2v[4];
    #pragma unroll
    for (int ct = 0; ct < 4; ++ct) b2v[ct] = b2[colbase + ct * 16 + m];
    #pragma unroll
    for (int rt = 0; rt < 2; ++rt) {
        #pragma unroll
        for (int r = 0; r < 8; ++r) {
            int row = rowbase + rt * 16 + r + kh * 8;  // C layout: vgpr r -> M=r (+8 for lanes 16-31)
            float a2v = a2[row];
            unsigned long long key = ~0ull;
            #pragma unroll
            for (int ct = 0; ct < 4; ++ct) {
                float d = fmaxf(a2v + b2v[ct] - 2.f * acc[rt][ct][r], 0.f);
                unsigned long long kk =
                    ((unsigned long long)__float_as_uint(d) << 32) | (unsigned)(colbase + ct * 16 + m);
                key = kk < key ? kk : key;
            }
            for (int off = 1; off < 16; off <<= 1) {   // reduce within each 16-lane group
                unsigned long long o = __shfl_xor(key, off, 32);
                key = o < key ? o : key;
            }
            if (m == 0) atomicMin(&smin[(wave >> 2) * 32 + rt * 16 + r + kh * 8], key); // ds_min_u64
        }
    }
    __syncthreads();
    if (tid < 64) atomicMin(&minpack[(size_t)blockIdx.y * 64 + tid], smin[tid]); // global_atomic_min_u64
}

// ---------- 4) per-row min -> minval[], argmax -> s_star, s_idx, m_star ----------
__global__ __launch_bounds__(256) void finalize_min_kernel(
    const unsigned long long* __restrict__ mp, float* __restrict__ minval,
    float* __restrict__ scalf, int* __restrict__ scali)
{
    int tid = threadIdx.x;
    float best = -1.f; int bidx = 0x7fffffff;
    for (int i = tid; i < NP; i += 256) {
        unsigned long long k = mp[i];
        float d = sqrtf(__uint_as_float((unsigned)(k >> 32)));
        minval[i] = d;
        if (d > best || (d == best && i < bidx)) { best = d; bidx = i; }
    }
    for (int off = 16; off > 0; off >>= 1) {
        float ov = __shfl_xor(best, off, 32);
        int   oi = __shfl_xor(bidx, off, 32);
        if (ov > best || (ov == best && oi < bidx)) { best = ov; bidx = oi; }
    }
    __shared__ float bv[8]; __shared__ int bi[8];
    if ((tid & 31) == 0) { bv[tid >> 5] = best; bi[tid >> 5] = bidx; }
    __syncthreads();
    if (tid == 0) {
        for (int i = 1; i < 8; ++i)
            if (bv[i] > best || (bv[i] == best && bi[i] < bidx)) { best = bv[i]; bidx = bi[i]; }
        scalf[0] = best;                                    // s_star
        scali[1] = bidx;                                    // s_idx
        scali[2] = (int)(mp[bidx] & 0xffffffffu);           // m_star lib index
    }
}

// ---------- 5) w_dist: one wave per lib row ----------
__global__ __launch_bounds__(256) void wdist_kernel(
    const float* __restrict__ lib, const float* __restrict__ b2,
    const int* __restrict__ scali, float* __restrict__ wd)
{
    int ms   = scali[2];
    int j    = blockIdx.x * 8 + (threadIdx.x >> 5);
    int lane = threadIdx.x & 31;
    const float* mrow = lib + (size_t)ms * DIM;
    const float* jrow = lib + (size_t)j * DIM;
    float dot = 0.f;
    for (int k = lane; k < DIM; k += 32) dot += jrow[k] * mrow[k];
    for (int off = 16; off > 0; off >>= 1) dot += __shfl_xor(dot, off, 32);
    if (lane == 0) {
        float d2 = fmaxf(b2[j] + b2[ms] - 2.f * dot, 0.f);
        wd[j] = sqrtf(d2);
    }
}

// ---------- 6) 3 smallest of w_dist, knn distances, final scalar s ----------
__global__ __launch_bounds__(256) void final_score_kernel(
    const float* __restrict__ wd, const float* __restrict__ lib,
    const float* __restrict__ patch, const float* __restrict__ mean,
    const float* __restrict__ std_, const float* __restrict__ scalf,
    const int* __restrict__ scali, float* __restrict__ out)
{
    __shared__ int   ex[3];
    __shared__ float redf[8]; __shared__ int redi[8];
    __shared__ float knn[2];
    int tid = threadIdx.x;
    for (int pass = 0; pass < 3; ++pass) {
        int e0 = pass > 0 ? ex[0] : -1;
        int e1 = pass > 1 ? ex[1] : -1;
        float best = 3.4e38f; int bidx = 0x7fffffff;
        for (int i = tid; i < NL; i += 256) {
            if (i == e0 || i == e1) continue;
            float v = wd[i];
            if (v < best || (v == best && i < bidx)) { best = v; bidx = i; }
        }
        for (int off = 16; off > 0; off >>= 1) {
            float ov = __shfl_xor(best, off, 32);
            int   oi = __shfl_xor(bidx, off, 32);
            if (ov < best || (ov == best && oi < bidx)) { best = ov; bidx = oi; }
        }
        if ((tid & 31) == 0) { redf[tid >> 5] = best; redi[tid >> 5] = bidx; }
        __syncthreads();
        if (tid == 0) {
            for (int i = 1; i < 8; ++i)
                if (redf[i] < best || (redf[i] == best && redi[i] < bidx)) { best = redf[i]; bidx = redi[i]; }
            ex[pass] = bidx;
        }
        __syncthreads();
    }
    // m_star_knn = || m_test - lib[nn_idx[1..2]] ||
    int sidx = scali[1];
    float mu = mean[0], sd = std_[0];
    const float* trow = patch + (size_t)sidx * DIM;
    for (int t = 0; t < 2; ++t) {
        const float* lrow = lib + (size_t)ex[t + 1] * DIM;
        float acc = 0.f;
        for (int k = tid; k < DIM; k += 256) {
            float diff = (trow[k] - mu) / sd - lrow[k];
            acc += diff * diff;
        }
        for (int off = 16; off > 0; off >>= 1) acc += __shfl_xor(acc, off, 32);
        if ((tid & 31) == 0) redf[tid >> 5] = acc;
        __syncthreads();
        if (tid == 0) {
            float s = 0.f; for (int i = 0; i < 8; ++i) s += redf[i];
            knn[t] = sqrtf(fmaxf(s, 0.f));
        }
        __syncthreads();
    }
    if (tid == 0) {
        float Ds = sqrtf((float)DIM);
        float s_star = scalf[0];
        float w = 1.f - expf(s_star / Ds) / (expf(knn[0] / Ds) + expf(knn[1] / Ds));
        out[0] = w * s_star;
    }
}

// ---------- 7) bilinear 56 -> 224 (half-pixel centers, edge clamp) ----------
__global__ __launch_bounds__(256) void resize_kernel(
    const float* __restrict__ mv, float* __restrict__ outimg)
{
    int idx = blockIdx.x * 256 + threadIdx.x;
    if (idx >= IMG * IMG) return;
    int y = idx / IMG, x = idx % IMG;
    float sy = (y + 0.5f) * ((float)FM / IMG) - 0.5f;
    float sx = (x + 0.5f) * ((float)FM / IMG) - 0.5f;
    float fy = sy - floorf(sy); int y0 = (int)floorf(sy);
    float fx = sx - floorf(sx); int x0 = (int)floorf(sx);
    int y1 = iclampi(y0 + 1, 0, FM - 1); y0 = iclampi(y0, 0, FM - 1);
    int x1 = iclampi(x0 + 1, 0, FM - 1); x0 = iclampi(x0, 0, FM - 1);
    float v00 = mv[y0 * FM + x0], v01 = mv[y0 * FM + x1];
    float v10 = mv[y1 * FM + x0], v11 = mv[y1 * FM + x1];
    float v0 = v00 + (v01 - v00) * fx;
    float v1 = v10 + (v11 - v10) * fx;
    outimg[idx] = v0 + (v1 - v0) * fy;
}

// ---------- 8) separable 9-tap Gaussian (sigma=4), reflect boundary ----------
__global__ __launch_bounds__(256) void blur_h_kernel(const float* __restrict__ in, float* __restrict__ out) {
    int idx = blockIdx.x * 256 + threadIdx.x;
    if (idx >= IMG * IMG) return;
    int y = idx / IMG, x = idx % IMG;
    float acc = 0.f, wsum = 0.f;
    for (int t = -4; t <= 4; ++t) {
        float g = expf(-(float)(t * t) / 32.f);
        acc += g * in[y * IMG + mirror(x + t, IMG)];
        wsum += g;
    }
    out[idx] = acc / wsum;
}
__global__ __launch_bounds__(256) void blur_v_kernel(const float* __restrict__ in, float* __restrict__ out) {
    int idx = blockIdx.x * 256 + threadIdx.x;
    if (idx >= IMG * IMG) return;
    int y = idx / IMG, x = idx % IMG;
    float acc = 0.f, wsum = 0.f;
    for (int t = -4; t <= 4; ++t) {
        float g = expf(-(float)(t * t) / 32.f);
        acc += g * in[mirror(y + t, IMG) * IMG + x];
        wsum += g;
    }
    out[idx] = acc / wsum;
}

// ---------- launch ----------
extern "C" void kernel_launch(void* const* d_in, const int* in_sizes, int n_in,
                              void* d_out, int out_size, void* d_ws, size_t ws_size,
                              hipStream_t stream) {
    (void)in_sizes; (void)n_in; (void)out_size; (void)ws_size;
    const float* patch = (const float*)d_in[0];
    const float* lib   = (const float*)d_in[1];
    const float* mean  = (const float*)d_in[2];
    const float* std_  = (const float*)d_in[3];
    float* out = (float*)d_out;

    char* ws = (char*)d_ws;
    const size_t SA = (size_t)NP * DIM * 2;   // bf16 plane of patch
    const size_t SB = (size_t)NL * DIM * 2;   // bf16 plane of lib
    unsigned short* Ahi = (unsigned short*)(ws);
    unsigned short* Alo = (unsigned short*)(ws + SA);
    unsigned short* Bhi = (unsigned short*)(ws + 2 * SA);
    unsigned short* Blo = (unsigned short*)(ws + 2 * SA + SB);
    char* p = ws + 2 * SA + 2 * SB;
    float* a2 = (float*)p;                    p += (size_t)NP * 4;
    float* b2 = (float*)p;                    p += (size_t)NL * 4;
    unsigned long long* minpack = (unsigned long long*)p; p += (size_t)NP * 8;
    float* minval = (float*)p;                p += (size_t)NP * 4;
    float* wd = (float*)p;                    p += (size_t)NL * 4;
    float* scalf = (float*)p; int* scali = (int*)p; p += 256;
    float* resized = (float*)p;               p += (size_t)IMG * IMG * 4;
    float* tmpH = (float*)p;                  p += (size_t)IMG * IMG * 4;

    prep_kernel<<<NP, 256, 0, stream>>>(patch, mean, std_, Ahi, Alo, a2, 1);
    prep_kernel<<<NL, 256, 0, stream>>>(lib,   mean, std_, Bhi, Blo, b2, 0);
    init_minpack<<<(NP + 255) / 256, 256, 0, stream>>>(minpack, NP);

    dim3 g(NL / 256, NP / 64);                // 64 x 49 blocks, 8 waves each
    gemm_min_kernel<<<g, 256, 0, stream>>>(Ahi, Alo, Bhi, Blo, a2, b2, minpack);

    finalize_min_kernel<<<1, 256, 0, stream>>>(minpack, minval, scalf, scali);
    wdist_kernel<<<NL / 8, 256, 0, stream>>>(lib, b2, scali, wd);
    final_score_kernel<<<1, 256, 0, stream>>>(wd, lib, patch, mean, std_, scalf, scali, out);

    const int npix = IMG * IMG;
    resize_kernel<<<(npix + 255) / 256, 256, 0, stream>>>(minval, resized);
    blur_h_kernel<<<(npix + 255) / 256, 256, 0, stream>>>(resized, tmpH);
    blur_v_kernel<<<(npix + 255) / 256, 256, 0, stream>>>(tmpH, out + 1);
}
